// SAN_decoder_62466004353389
// MI455X (gfx1250) — matrix operand order repaired and checked by reading, not compiled, not run on hardware
//
#include <hip/hip_runtime.h>

// ---------------- types / helpers ----------------
typedef __attribute__((ext_vector_type(16))) __bf16 v16bf;
typedef __attribute__((ext_vector_type(8)))  __bf16 v8bf;
typedef __attribute__((ext_vector_type(8)))  float  v8f;

#define LB256 __launch_bounds__(256, 1)

__device__ __forceinline__ v8f vzero() {
  v8f z = {0.f,0.f,0.f,0.f,0.f,0.f,0.f,0.f};
  return z;
}

// A-fragment (16xK row-major bf16, ld = ldK elements). Contiguous k order (matches B):
// lanes 0-15 : row=lane, K [0,16) of the 32-wide k-step
// lanes 16-31: row=lane-16, K [16,32)
__device__ __forceinline__ v16bf load_a_frag(const __bf16* base, int ldK, int kbase) {
  int lane = threadIdx.x & 31;
  const __bf16* p0 = base + (long)(lane & 15) * ldK + kbase + ((lane >> 4) << 4);
  v8bf lo = *(const v8bf*)p0;
  v8bf hi = *(const v8bf*)(p0 + 8);
  v16bf r;
#pragma unroll
  for (int e = 0; e < 8; ++e) { r[e] = lo[e]; r[8 + e] = hi[e]; }
  return r;
}

// B-fragment from (N x K) row-major bf16 weights: lane -> N = nbase+(lane&15),
// K = kbase + (lane>>4)*16 .. +16 contiguous -> single 32B load.
__device__ __forceinline__ v16bf load_b_frag(const __bf16* W, int ldK, int nbase, int kbase) {
  int lane = threadIdx.x & 31;
  return *(const v16bf*)(W + (long)(nbase + (lane & 15)) * ldK + kbase + ((lane >> 4) << 4));
}

__device__ __forceinline__ void mma(v8f& c, v16bf a, v16bf b) {
  c = __builtin_amdgcn_wmma_f32_16x16x32_bf16(false, a, false, b, (short)0, c, false, false);
}

__device__ __forceinline__ float sig_(float x)  { return 1.f / (1.f + __expf(-x)); }
__device__ __forceinline__ float tanh_(float x) { return 2.f * sig_(2.f * x) - 1.f; }

// ---------------- problem dims ----------------
constexpr int  Bn = 16, Tn = 128, Cc = 684, An = 512;
constexpr int  KG1 = 512, KCI = 768, KG2 = 960, KPW = 1216, KPC = 1472, KFT = 704;
constexpr long FEAT_B = (long)Cc * 1024;           // 700416 per batch
constexpr long OUT_WP = 0;
constexpr long OUT_SP = 524288;
constexpr long OUT_WA = 538624;
constexpr long OUT_CP = 2635776;
constexpr long OUT_CA = 3160064;

// ---------------- mega pointer pack ----------------
struct Ptrs {
  // inputs
  const float* feat; const int* labels; const float* imask;
  const float* emb; const float* init_w; const float* init_b;
  const float* g1_wi;  const float* g1_wh;  const float* g1_bi;  const float* g1_bh;
  const float* g2w_wi; const float* g2w_wh; const float* g2w_bi; const float* g2w_bh;
  const float* gci_wi; const float* gci_wh; const float* gci_bi; const float* gci_bh;
  const float* g2c_wi; const float* g2c_wh; const float* g2c_bi; const float* g2c_bh;
  const float* wa_encw; const float* wa_encb; const float* wa_hidw; const float* wa_hidb;
  const float* wa_convk; const float* wa_attw; const float* wa_alphaw; const float* wa_alphab;
  const float* ca_encw; const float* ca_encb; const float* ca_hidw; const float* ca_hidb;
  const float* ca_convk; const float* ca_attw; const float* ca_alphaw; const float* ca_alphab;
  const float* wsw_w; const float* wsw_b; const float* wew_w; const float* wew_b;
  const float* wcw_w; const float* wcw_b; const float* wconv_w; const float* wconv_b;
  const float* sconv_w; const float* sconv_b;
  const float* csw_w; const float* csw_b; const float* cww_w; const float* cww_b;
  const float* crw_w; const float* crw_b; const float* ccw_w; const float* ccw_b;
  const float* cconv_w; const float* cconv_b;
  // workspace bf16
  __bf16 *featT, *featW, *featC, *KpW, *KpC, *Wg1, *Wci, *Wg2;
  __bf16 *WhidW, *WhidC, *WencW, *WencC, *WprojW, *WprojC, *WoutW, *WoutC, *h1b, *c1b;
  // workspace f32
  float *biasPW, *biasPC, *ph, *was, *c2ph, *c2psum, *m2d, *avg, *h1f;
  float *qW, *qC, *eW, *eC, *ctxW, *ctxC;
  float* out;
};

// ---------------- precompute: pack weights to bf16 WMMA layouts ----------------
__global__ void k_pack(Ptrs P) {
  int job = blockIdx.z;
  long idx = (long)blockIdx.x * 256 + threadIdx.x;
  switch (job) {
    case 0: { // Wg1 1024x512: rows 0-511 [wi|wh] (r,z), 512-767 [wi_n|0], 768-1023 [0|wh_n]
      if (idx >= 1024L * 512) return;
      int n = (int)(idx >> 9), kc = (int)(idx & 511); float v = 0.f;
      if (n < 512)      v = (kc < 256) ? P.g1_wi[n*256+kc] : P.g1_wh[n*256+kc-256];
      else if (n < 768) { if (kc < 256) v = P.g1_wi[n*256+kc]; }
      else              { if (kc >= 256) v = P.g1_wh[(n-256)*256+kc-256]; }
      P.Wg1[idx] = (__bf16)v; } break;
    case 1: { // Wci 1024x768
      if (idx >= 1024L * 768) return;
      int n = (int)(idx / 768), kc = (int)(idx % 768); float v = 0.f;
      if (n < 512)      v = (kc < 512) ? P.gci_wi[n*512+kc] : P.gci_wh[n*256+kc-512];
      else if (n < 768) { if (kc < 512) v = P.gci_wi[n*512+kc]; }
      else              { if (kc >= 512) v = P.gci_wh[(n-256)*256+kc-512]; }
      P.Wci[idx] = (__bf16)v; } break;
    case 2: { // Wg2 2048x960: [gru_wo ; gru_co], K = [ctx 684 | h 256 | pad 20]
      if (idx >= 2048L * 960) return;
      int n = (int)(idx / 960), kc = (int)(idx % 960);
      int jb = n >> 10, nn = n & 1023;
      const float* wi = jb ? P.g2c_wi : P.g2w_wi;
      const float* wh = jb ? P.g2c_wh : P.g2w_wh;
      float v = 0.f;
      if (nn < 512)      { if (kc < 684) v = wi[nn*684+kc]; else if (kc < 940) v = wh[nn*256+kc-684]; }
      else if (nn < 768) { if (kc < 684) v = wi[nn*684+kc]; }
      else               { if (kc >= 684 && kc < 940) v = wh[(nn-256)*256+kc-684]; }
      P.Wg2[idx] = (__bf16)v; } break;
    case 3: { if (idx >= 512L*256) return; P.WhidW[idx] = (__bf16)P.wa_hidw[idx]; } break;
    case 4: { if (idx >= 512L*256) return; P.WhidC[idx] = (__bf16)P.ca_hidw[idx]; } break;
    case 5: { // WprojW 128x1216: [wsw|wew|wcw|pad]
      if (idx >= 128L * 1216) return;
      int n = (int)(idx / 1216), kc = (int)(idx % 1216); float v = 0.f;
      if (kc < 256) v = P.wsw_w[n*256+kc];
      else if (kc < 512) v = P.wew_w[n*256+kc-256];
      else if (kc < 1196) v = P.wcw_w[n*684+kc-512];
      P.WprojW[idx] = (__bf16)v; } break;
    case 6: { // WprojC 128x1472: [csw|cww|crw|ccw|pad]
      if (idx >= 128L * 1472) return;
      int n = (int)(idx / 1472), kc = (int)(idx % 1472); float v = 0.f;
      if (kc < 256) v = P.csw_w[n*256+kc];
      else if (kc < 512) v = P.cww_w[n*256+kc-256];
      else if (kc < 768) v = P.crw_w[n*256+kc-512];
      else if (kc < 1452) v = P.ccw_w[n*684+kc-768];
      P.WprojC[idx] = (__bf16)v; } break;
    case 7: { // WoutW 272x128: rows 0-255 wconv, 256-262 sconv, rest 0
      if (idx >= 272L * 128) return;
      int n = (int)(idx >> 7), kc = (int)(idx & 127); float v = 0.f;
      if (n < 256) v = P.wconv_w[n*128+kc];
      else if (n < 263) v = P.sconv_w[(n-256)*128+kc];
      P.WoutW[idx] = (__bf16)v; } break;
    case 8: { if (idx >= 256L*128) return; P.WoutC[idx] = (__bf16)P.cconv_w[idx]; } break;
    case 9: { // WencW 512x704 (K-pad 684->704)
      if (idx >= 512L * 704) return;
      int n = (int)(idx / 704), kc = (int)(idx % 704);
      P.WencW[idx] = (__bf16)((kc < 684) ? P.wa_encw[n*684+kc] : 0.f); } break;
    case 10: {
      if (idx >= 512L * 704) return;
      int n = (int)(idx / 704), kc = (int)(idx % 704);
      P.WencC[idx] = (__bf16)((kc < 684) ? P.ca_encw[n*684+kc] : 0.f); } break;
    case 11: case 12: { // fused conv kernels Kp[a][128] = att_w @ conv_k  (121 taps, pad 128)
      if (idx >= 512L * 128) return;
      int a = (int)(idx >> 7), k = (int)(idx & 127); float v = 0.f;
      if (k < 121) {
        const float* aw = (job == 11) ? P.wa_attw : P.ca_attw;
        const float* ck = (job == 11) ? P.wa_convk : P.ca_convk;
        for (int c = 0; c < 512; ++c) v += aw[a*512+c] * ck[c*121+k];
      }
      ((job == 11) ? P.KpW : P.KpC)[idx] = (__bf16)v; } break;
    case 13: { // combined projection biases
      if (idx < 128)      P.biasPW[idx] = P.wsw_b[idx] + P.wew_b[idx] + P.wcw_b[idx];
      else if (idx < 256) { int n = (int)idx - 128;
                            P.biasPC[n] = P.csw_b[n] + P.cww_b[n] + P.crw_b[n] + P.ccw_b[n]; }
    } break;
  }
}

// ---------------- precompute: feature transpose (b,c,p)->(bp,c) bf16 ----------------
__global__ void k_featT(Ptrs P) {
  int bp = blockIdx.x;                  // 0..16383
  int b = bp >> 10, p = bp & 1023;
  for (int kc = threadIdx.x; kc < KFT; kc += 256) {
    float v = (kc < Cc) ? P.feat[(long)b * FEAT_B + (long)kc * 1024 + p] : 0.f;
    P.featT[(long)bp * KFT + kc] = (__bf16)v;
  }
}

// ---------------- precompute: feat_trans = featT @ enc_w^T + enc_b  (WMMA) ----------------
// grid (128, 4, 2): 8 N-tiles per block -> 8 accumulator tiles per wave (no spills)
__global__ void LB256 k_featproj(Ptrs P) {
  int mg = blockIdx.x, nh = blockIdx.y, job = blockIdx.z;
  int wv = threadIdx.x >> 5, lane = threadIdx.x & 31;
  int mbase = mg * 128 + wv * 16;       // global position row
  const __bf16* Wenc = job ? P.WencC : P.WencW;
  const float*  encb = job ? P.ca_encb : P.wa_encb;
  __bf16* dst = job ? P.featC : P.featW;
  v8f acc[8];
#pragma unroll
  for (int t = 0; t < 8; ++t) acc[t] = vzero();
  for (int ks = 0; ks < KFT / 32; ++ks) {
    v16bf a = load_a_frag(P.featT + (long)mbase * KFT, KFT, ks * 32);
#pragma unroll
    for (int t = 0; t < 8; ++t)
      mma(acc[t], a, load_b_frag(Wenc, KFT, (nh * 8 + t) * 16, ks * 32));
  }
  int sel = lane >> 4, nl = lane & 15;
#pragma unroll
  for (int t = 0; t < 8; ++t) {
    int n = (nh * 8 + t) * 16 + nl;
    float eb = encb[n];
#pragma unroll
    for (int r = 0; r < 8; ++r) {
      int row = mbase + sel * 8 + r;
      dst[(long)row * 512 + n] = (__bf16)(acc[t][r] + eb);
    }
  }
}

// ---------------- precompute: mask2d + masked average ----------------
__global__ void k_avg(Ptrs P) {
  __shared__ float m2dl[1024];
  __shared__ float red[256];
  int b = blockIdx.x, t = threadIdx.x;
  for (int p = t; p < 1024; p += 256) {
    int him = (p >> 6) * 16, wim = (p & 63) * 16;
    float m = P.imask[(long)b * 262144 + (long)him * 1024 + wim];
    m2dl[p] = m; P.m2d[b * 1024 + p] = m;
  }
  __syncthreads();
  float s = 0.f;
  for (int p = t; p < 1024; p += 256) s += m2dl[p];
  red[t] = s; __syncthreads();
  for (int o = 128; o; o >>= 1) { if (t < o) red[t] += red[t + o]; __syncthreads(); }
  float inv = 1.f / red[0];
  for (int c = t; c < Cc; c += 256) {
    const float* fr = P.feat + (long)b * FEAT_B + (long)c * 1024;
    float a = 0.f;
    for (int p = 0; p < 1024; ++p) a += fr[p] * m2dl[p];
    P.avg[b * Cc + c] = a * inv;
  }
}

__global__ void k_init(Ptrs P) {
  int b = blockIdx.x, u = threadIdx.x;
  if (u >= 256) return;
  float a = P.init_b[u];
  const float* av = P.avg + b * Cc;
  const float* w = P.init_w + u * Cc;
  for (int c = 0; c < Cc; ++c) a += w[c] * av[c];
  P.ph[b * 256 + u] = tanh_(a);
}

// ---------------- per-step: GRU1 (h1) and GRU_ci (c1), fused gate GEMMs ----------------
__global__ void LB256 k_gru1(Ptrs P, int step) {
  int job = blockIdx.x;                     // 0: h1, 1: c1
  __shared__ alignas(16) __bf16 xcat[16 * KCI];
  __shared__ int tokS[16], parS[16], chS[16], relS[16];
  int t = threadIdx.x;
  if (t < 16) {
    int b = t;
    if (job == 0) {
      const int* L = P.labels + b * 512 + step * 4;
      tokS[b] = L[3]; parS[b] = L[2] * 16 + b;
    } else {
      const int* L = P.labels + b * 512 + (Tn - 1 - step) * 4;
      int child = L[1], rel0 = L[3];
      bool instr = (rel0 >= 4 && rel0 <= 10);
      int rel = (child == 2) ? 2 : ((!instr && rel0 != 1) ? 3 : rel0);
      chS[b] = child; relS[b] = rel;
    }
  }
  __syncthreads();
  int KX = job ? KCI : KG1;
  for (int idx = t; idx < 16 * KX; idx += 256) {
    int b = idx / KX, kc = idx % KX; float v;
    if (job == 0)
      v = (kc < 256) ? P.emb[tokS[b] * 256 + kc] : P.ph[parS[b] * 256 + kc - 256];
    else
      v = (kc < 256) ? P.emb[chS[b] * 256 + kc]
        : (kc < 512) ? P.emb[relS[b] * 256 + kc - 256]
                     : P.c2ph[b * 256 + kc - 512];
    xcat[b * KX + kc] = (__bf16)v;
  }
  __syncthreads();
  const __bf16* W = job ? P.Wci : P.Wg1;
  int wv = t >> 5, lane = t & 31, sel = lane >> 4, nl = lane & 15;
  v8f aR[2], aZ[2], aGi[2], aGh[2];
#pragma unroll
  for (int q = 0; q < 2; ++q) { aR[q]=vzero(); aZ[q]=vzero(); aGi[q]=vzero(); aGh[q]=vzero(); }
  for (int ks = 0; ks < KX / 32; ++ks) {
    v16bf af = load_a_frag(xcat, KX, ks * 32);
#pragma unroll
    for (int q = 0; q < 2; ++q) {
      int k = 2 * wv + q;
      mma(aR[q],  af, load_b_frag(W, KX, (k)      * 16, ks * 32));
      mma(aZ[q],  af, load_b_frag(W, KX, (16 + k) * 16, ks * 32));
      mma(aGi[q], af, load_b_frag(W, KX, (32 + k) * 16, ks * 32));
      mma(aGh[q], af, load_b_frag(W, KX, (48 + k) * 16, ks * 32));
    }
  }
  const float* bi = job ? P.gci_bi : P.g1_bi;
  const float* bh = job ? P.gci_bh : P.g1_bh;
#pragma unroll
  for (int q = 0; q < 2; ++q) {
    int u = (2 * wv + q) * 16 + nl;
    float biR = bi[u], bhR = bh[u], biZ = bi[256+u], bhZ = bh[256+u];
    float biN = bi[512+u], bhN = bh[512+u];
#pragma unroll
    for (int r = 0; r < 8; ++r) {
      int b = sel * 8 + r;
      float rr = sig_(aR[q][r] + biR + bhR);
      float zz = sig_(aZ[q][r] + biZ + bhZ);
      float nn = tanh_((aGi[q][r] + biN) + rr * (aGh[q][r] + bhN));
      float hp = job ? P.c2ph[b * 256 + u] : P.ph[parS[b] * 256 + u];
      float h = (1.f - zz) * nn + zz * hp;
      if (job == 0) { P.h1f[b*256+u] = h; P.h1b[b*256+u] = (__bf16)h; }
      else          { P.c1b[b*256+u] = (__bf16)h; }
    }
  }
}

// ---------------- per-step: attention queries (WMMA) ----------------
__global__ void LB256 k_query(Ptrs P) {
  int job = blockIdx.x;
  const __bf16* A = job ? P.c1b : P.h1b;
  const __bf16* W = job ? P.WhidC : P.WhidW;
  const float* bias = job ? P.ca_hidb : P.wa_hidb;
  float* q = job ? P.qC : P.qW;
  int wv = threadIdx.x >> 5, lane = threadIdx.x & 31, sel = lane >> 4, nl = lane & 15;
  v8f acc[4];
#pragma unroll
  for (int j = 0; j < 4; ++j) acc[j] = vzero();
  for (int ks = 0; ks < 8; ++ks) {
    v16bf af = load_a_frag(A, 256, ks * 32);
#pragma unroll
    for (int j = 0; j < 4; ++j)
      mma(acc[j], af, load_b_frag(W, 256, (wv * 4 + j) * 16, ks * 32));
  }
#pragma unroll
  for (int j = 0; j < 4; ++j) {
    int n = (wv * 4 + j) * 16 + nl;
    float bv = bias[n];
#pragma unroll
    for (int r = 0; r < 8; ++r) q[(sel * 8 + r) * 512 + n] = acc[j][r] + bv;
  }
}

// ---------------- per-step: attention energy (fused conv-as-GEMM, WMMA) ----------------
__global__ void LB256 k_energy(Ptrs P, int step) {
  int b = blockIdx.x, pg = blockIdx.y, job = blockIdx.z;
  int pbase = pg * 128;
  __shared__ alignas(16) __bf16 patch[128 * 128];   // im2col patches, K padded 121->128
  __shared__ float asum_l[1024];
  __shared__ float e_l[128];
  int t = threadIdx.x;
  const float* asrc = job ? (P.c2psum + b * 1024)
                          : (P.was + (long)(P.labels[b*512 + step*4 + 2] * 16 + b) * 1024);
  for (int p = t; p < 1024; p += 256) asum_l[p] = asrc[p];
  __syncthreads();
  for (int idx = t; idx < 128 * 128; idx += 256) {
    int pos = idx >> 7, k = idx & 127; float v = 0.f;
    if (k < 121) {
      int ii = k / 11, jj = k - ii * 11;
      int p = pbase + pos;
      int hh = (p >> 6) + ii - 5, ww = (p & 63) + jj - 5;
      if (hh >= 0 && hh < 16 && ww >= 0 && ww < 64) v = asum_l[hh * 64 + ww];
    }
    patch[idx] = (__bf16)v;
  }
  __syncthreads();
  const __bf16* Kp = job ? P.KpC : P.KpW;
  const float* q  = (job ? P.qC : P.qW) + b * 512;
  const float* aw = job ? P.ca_alphaw : P.wa_alphaw;
  const __bf16* ft = (job ? P.featC : P.featW) + (long)(b * 1024 + pbase) * 512;
  int wv = t >> 5, lane = t & 31, sel = lane >> 4, nl = lane & 15;
  float eacc[8] = {0,0,0,0,0,0,0,0};
  for (int nt = 0; nt < 32; ++nt) {
    v8f acc = vzero();
#pragma unroll
    for (int ks = 0; ks < 4; ++ks)
      mma(acc, load_a_frag(patch + wv * 16 * 128, 128, ks * 32),
               load_b_frag(Kp, 128, nt * 16, ks * 32));
    int a = nt * 16 + nl;
    float qv = q[a], av = aw[a];
#pragma unroll
    for (int r = 0; r < 8; ++r) {
      int pl = wv * 16 + sel * 8 + r;
      float x = acc[r] + qv + (float)ft[(long)pl * 512 + a];
      eacc[r] += av * tanh_(x);
    }
  }
#pragma unroll
  for (int m = 1; m <= 8; m <<= 1)
#pragma unroll
    for (int r = 0; r < 8; ++r) eacc[r] += __shfl_xor(eacc[r], m, 32);
  if (nl == 0)
#pragma unroll
    for (int r = 0; r < 8; ++r) e_l[wv * 16 + sel * 8 + r] = eacc[r];
  __syncthreads();
  float ab = *(job ? P.ca_alphab : P.wa_alphab);
  float* eo = (job ? P.eC : P.eW) + b * 1024 + pbase;
  if (t < 128) eo[t] = e_l[t] + ab;
}

// ---------------- per-step: softmax + alpha + alpha_sum update ----------------
__global__ void k_softmax(Ptrs P, int step) {
  int b = blockIdx.x, job = blockIdx.y, t = threadIdx.x;
  __shared__ float red[256];
  const float* e = (job ? P.eC : P.eW) + b * 1024;
  float mx = -1e30f;
  for (int p = t; p < 1024; p += 256) mx = fmaxf(mx, e[p]);
  red[t] = mx; __syncthreads();
  for (int o = 128; o; o >>= 1) { if (t < o) red[t] = fmaxf(red[t], red[t+o]); __syncthreads(); }
  float m = red[0]; __syncthreads();
  float ex[4]; float s = 0.f;
#pragma unroll
  for (int k = 0; k < 4; ++k) {
    int p = t + k * 256;
    ex[k] = __expf(e[p] - m) * P.m2d[b * 1024 + p];
    s += ex[k];
  }
  red[t] = s; __syncthreads();
  for (int o = 128; o; o >>= 1) { if (t < o) red[t] += red[t+o]; __syncthreads(); }
  float tot = red[0] + 1e-10f;
  float* ao = P.out + (job ? OUT_CA + ((long)b * Tn + (Tn - 1 - step)) * 1024
                           : OUT_WA + ((long)b * Tn + step) * 1024);
  if (job == 0) {
    int par = P.labels[b*512 + step*4 + 2] * 16 + b;
    int nw = (step + 1) * 16 + b;
#pragma unroll
    for (int k = 0; k < 4; ++k) {
      int p = t + k * 256;
      float al = ex[k] / tot;
      ao[p] = al;
      P.was[(long)nw * 1024 + p] = al + P.was[(long)par * 1024 + p];
    }
  } else {
#pragma unroll
    for (int k = 0; k < 4; ++k) {
      int p = t + k * 256;
      float al = ex[k] / tot;
      ao[p] = al;
      P.c2psum[b * 1024 + p] = al + P.c2psum[b * 1024 + p];
    }
  }
}

// ---------------- per-step: context = alpha . feat ----------------
__global__ void k_ctx(Ptrs P, int step) {
  int b = blockIdx.x, cg = blockIdx.y, job = blockIdx.z;
  int wv = threadIdx.x >> 5, lane = threadIdx.x & 31;
  int c = cg * 8 + wv;
  if (c >= Cc) return;
  const float* al = P.out + (job ? OUT_CA + ((long)b * Tn + (Tn - 1 - step)) * 1024
                                 : OUT_WA + ((long)b * Tn + step) * 1024);
  const float* fr = P.feat + (long)b * FEAT_B + (long)c * 1024;
  float s = 0.f;
  for (int p = lane; p < 1024; p += 32) {
    if (p + 256 < 1024) __builtin_prefetch(&fr[p + 256], 0, 1);
    s += al[p] * fr[p];
  }
#pragma unroll
  for (int m = 16; m; m >>= 1) s += __shfl_xor(s, m, 32);
  if (lane == 0) (job ? P.ctxC : P.ctxW)[b * Cc + c] = s;
}

// ---------------- per-step: GRU_wo / GRU_co (one block each, N=1024 GEMM) ----------------
__global__ void LB256 k_gru2(Ptrs P, int step) {
  int job = blockIdx.x;                     // 0: gru_wo -> hidden, 1: gru_co -> c2ph
  __shared__ alignas(16) __bf16 xcat[16 * KG2];     // [ctx_w 684 | h1 256 | pad 20]
  int t = threadIdx.x;
  for (int idx = t; idx < 16 * KG2; idx += 256) {
    int b = idx / KG2, kc = idx % KG2;
    float v = (kc < 684) ? P.ctxW[b * Cc + kc]
            : (kc < 940) ? P.h1f[b * 256 + kc - 684] : 0.f;
    xcat[idx] = (__bf16)v;
  }
  __syncthreads();
  int wv = t >> 5, lane = t & 31, sel = lane >> 4, nl = lane & 15;
  int nb = job * 64;
  v8f aR[2], aZ[2], aGi[2], aGh[2];
#pragma unroll
  for (int q = 0; q < 2; ++q) { aR[q]=vzero(); aZ[q]=vzero(); aGi[q]=vzero(); aGh[q]=vzero(); }
  for (int ks = 0; ks < KG2 / 32; ++ks) {
    v16bf af = load_a_frag(xcat, KG2, ks * 32);
#pragma unroll
    for (int q = 0; q < 2; ++q) {
      int k = 2 * wv + q;
      mma(aR[q],  af, load_b_frag(P.Wg2, KG2, (nb + k)      * 16, ks * 32));
      mma(aZ[q],  af, load_b_frag(P.Wg2, KG2, (nb + 16 + k) * 16, ks * 32));
      mma(aGi[q], af, load_b_frag(P.Wg2, KG2, (nb + 32 + k) * 16, ks * 32));
      mma(aGh[q], af, load_b_frag(P.Wg2, KG2, (nb + 48 + k) * 16, ks * 32));
    }
  }
  const float* bi = job ? P.g2c_bi : P.g2w_bi;
  const float* bh = job ? P.g2c_bh : P.g2w_bh;
#pragma unroll
  for (int q = 0; q < 2; ++q) {
    int u = (2 * wv + q) * 16 + nl;
    float biR = bi[u], bhR = bh[u], biZ = bi[256+u], bhZ = bh[256+u];
    float biN = bi[512+u], bhN = bh[512+u];
#pragma unroll
    for (int r = 0; r < 8; ++r) {
      int b = sel * 8 + r;
      float rr = sig_(aR[q][r] + biR + bhR);
      float zz = sig_(aZ[q][r] + biZ + bhZ);
      float nn = tanh_((aGi[q][r] + biN) + rr * (aGh[q][r] + bhN));
      float hp = P.h1f[b * 256 + u];
      float h = (1.f - zz) * nn + zz * hp;
      if (job == 0) P.ph[(long)((step + 1) * 16 + b) * 256 + u] = h;   // hidden
      else          P.c2ph[b * 256 + u] = h;                           // new c2ph
    }
  }
}

// ---------------- per-step: projections + vocab heads (two chained WMMA GEMMs) ----------------
__global__ void LB256 k_proj(Ptrs P, int step) {
  int job = blockIdx.x;                       // 0: word branch, 1: c2p branch
  __shared__ alignas(16) __bf16 xcat[16 * KPC];
  __shared__ alignas(16) __bf16 out1[16 * 128];
  __shared__ int tokS[16], chS[16], relS[16];
  int t = threadIdx.x;
  if (t < 16) {
    int b = t;
    if (job == 0) {
      tokS[b] = P.labels[b * 512 + step * 4 + 3];
    } else {
      const int* L = P.labels + b * 512 + (Tn - 1 - step) * 4;
      int child = L[1], rel0 = L[3];
      bool instr = (rel0 >= 4 && rel0 <= 10);
      int rel = (child == 2) ? 2 : ((!instr && rel0 != 1) ? 3 : rel0);
      chS[b] = child; relS[b] = rel;
    }
  }
  __syncthreads();
  int KX = job ? KPC : KPW;
  for (int idx = t; idx < 16 * KX; idx += 256) {
    int b = idx / KX, kc = idx % KX; float v;
    if (job == 0)
      v = (kc < 256)  ? P.ph[(long)((step + 1) * 16 + b) * 256 + kc]
        : (kc < 512)  ? P.emb[tokS[b] * 256 + kc - 256]
        : (kc < 1196) ? P.ctxW[b * Cc + kc - 512] : 0.f;
    else
      v = (kc < 256)  ? P.c2ph[b * 256 + kc]
        : (kc < 512)  ? P.emb[chS[b] * 256 + kc - 256]
        : (kc < 768)  ? P.emb[relS[b] * 256 + kc - 512]
        : (kc < 1452) ? P.ctxC[b * Cc + kc - 768] : 0.f;
    xcat[b * KX + kc] = (__bf16)v;
  }
  __syncthreads();
  int wv = t >> 5, lane = t & 31, sel = lane >> 4, nl = lane & 15;
  // GEMM1: M=16, N=128, K=KX
  const __bf16* W1 = job ? P.WprojC : P.WprojW;
  v8f acc = vzero();
  for (int ks = 0; ks < KX / 32; ++ks)
    mma(acc, load_a_frag(xcat, KX, ks * 32), load_b_frag(W1, KX, wv * 16, ks * 32));
  {
    int n = wv * 16 + nl;
    float bias = (job ? P.biasPC : P.biasPW)[n];
#pragma unroll
    for (int r = 0; r < 8; ++r) out1[(sel * 8 + r) * 128 + n] = (__bf16)(acc[r] + bias);
  }
  __syncthreads();
  // GEMM2: M=16, N=272 (word: wconv+sconv) or 256 (cconv), K=128
  const __bf16* W2 = job ? P.WoutC : P.WoutW;
  int NT2 = job ? 16 : 17;
  for (int nt = wv; nt < NT2; nt += 8) {
    v8f a2 = vzero();
#pragma unroll
    for (int ks = 0; ks < 4; ++ks)
      mma(a2, load_a_frag(out1, 128, ks * 32), load_b_frag(W2, 128, nt * 16, ks * 32));
    int n2 = nt * 16 + nl;
#pragma unroll
    for (int r = 0; r < 8; ++r) {
      int b = sel * 8 + r;
      if (job == 0) {
        if (n2 < 256)
          P.out[OUT_WP + ((long)b * Tn + step) * 256 + n2] = a2[r] + P.wconv_b[n2];
        else if (n2 < 263)
          P.out[OUT_SP + ((long)b * Tn + step) * 7 + (n2 - 256)] = a2[r] + P.sconv_b[n2 - 256];
      } else {
        P.out[OUT_CP + ((long)b * Tn + (Tn - 1 - step)) * 256 + n2] = a2[r] + P.cconv_b[n2];
      }
    }
  }
}

// ---------------- host orchestration ----------------
extern "C" void kernel_launch(void* const* d_in, const int* in_sizes, int n_in,
                              void* d_out, int out_size, void* d_ws, size_t ws_size,
                              hipStream_t stream) {
  (void)in_sizes; (void)n_in; (void)out_size;
  Ptrs P;
  const float** F = (const float**)d_in;
  P.feat = F[0]; P.labels = (const int*)d_in[1]; P.imask = F[2];
  P.emb = F[4]; P.init_w = F[5]; P.init_b = F[6];
  P.g1_wi = F[7];  P.g1_wh = F[8];  P.g1_bi = F[9];  P.g1_bh = F[10];
  P.g2w_wi = F[11]; P.g2w_wh = F[12]; P.g2w_bi = F[13]; P.g2w_bh = F[14];
  P.gci_wi = F[15]; P.gci_wh = F[16]; P.gci_bi = F[17]; P.gci_bh = F[18];
  P.g2c_wi = F[19]; P.g2c_wh = F[20]; P.g2c_bi = F[21]; P.g2c_bh = F[22];
  P.wa_encw = F[23]; P.wa_encb = F[24]; P.wa_hidw = F[25]; P.wa_hidb = F[26];
  P.wa_convk = F[27]; P.wa_attw = F[28]; P.wa_alphaw = F[29]; P.wa_alphab = F[30];
  P.ca_encw = F[31]; P.ca_encb = F[32]; P.ca_hidw = F[33]; P.ca_hidb = F[34];
  P.ca_convk = F[35]; P.ca_attw = F[36]; P.ca_alphaw = F[37]; P.ca_alphab = F[38];
  P.wsw_w = F[39]; P.wsw_b = F[40]; P.wew_w = F[41]; P.wew_b = F[42];
  P.wcw_w = F[43]; P.wcw_b = F[44]; P.wconv_w = F[45]; P.wconv_b = F[46];
  P.sconv_w = F[47]; P.sconv_b = F[48];
  P.csw_w = F[49]; P.csw_b = F[50]; P.cww_w = F[51]; P.cww_b = F[52];
  P.crw_w = F[53]; P.crw_b = F[54]; P.ccw_w = F[55]; P.ccw_b = F[56];
  P.cconv_w = F[57]; P.cconv_b = F[58];
  P.out = (float*)d_out;

  char* base = (char*)d_ws;
  size_t off = 0;
  auto carve = [&](size_t bytes) -> void* {
    void* p = base + off;
    off = (off + bytes + 255) & ~(size_t)255;
    return p;
  };
  P.featT  = (__bf16*)carve(16384L * KFT * 2);
  P.featW  = (__bf16*)carve(16384L * 512 * 2);
  P.featC  = (__bf16*)carve(16384L * 512 * 2);
  P.KpW    = (__bf16*)carve(512L * 128 * 2);
  P.KpC    = (__bf16*)carve(512L * 128 * 2);
  P.Wg1    = (__bf16*)carve(1024L * KG1 * 2);
  P.Wci    = (__bf16*)carve(1024L * KCI * 2);
  P.Wg2    = (__bf16*)carve(2048L * KG2 * 2);
  P.WhidW  = (__bf16*)carve(512L * 256 * 2);
  P.WhidC  = (__bf16*)carve(512L * 256 * 2);
  P.WencW  = (__bf16*)carve(512L * KFT * 2);
  P.WencC  = (__bf16*)carve(512L * KFT * 2);
  P.WprojW = (__bf16*)carve(128L * KPW * 2);
  P.WprojC = (__bf16*)carve(128L * KPC * 2);
  P.WoutW  = (__bf16*)carve(272L * 128 * 2);
  P.WoutC  = (__bf16*)carve(256L * 128 * 2);
  P.h1b    = (__bf16*)carve(16L * 256 * 2);
  P.c1b    = (__bf16*)carve(16L * 256 * 2);
  P.biasPW = (float*)carve(128 * 4);
  P.biasPC = (float*)carve(128 * 4);
  size_t phBytes  = 129L * 16 * 256 * 4;
  size_t wasBytes = 129L * 16 * 1024 * 4;
  P.ph     = (float*)carve(phBytes);
  P.was    = (float*)carve(wasBytes);
  P.c2ph   = (float*)carve(16L * 256 * 4);
  P.c2psum = (float*)carve(16L * 1024 * 4);
  P.m2d    = (float*)carve(16L * 1024 * 4);
  P.avg    = (float*)carve(16L * Cc * 4);
  P.h1f    = (float*)carve(16L * 256 * 4);
  P.qW     = (float*)carve(16L * 512 * 4);
  P.qC     = (float*)carve(16L * 512 * 4);
  P.eW     = (float*)carve(16L * 1024 * 4);
  P.eC     = (float*)carve(16L * 1024 * 4);
  P.ctxW   = (float*)carve(16L * Cc * 4);
  P.ctxC   = (float*)carve(16L * Cc * 4);
  (void)ws_size;

  // zero recurrent state (graph replays must start identical)
  hipMemsetAsync(P.ph, 0, phBytes, stream);
  hipMemsetAsync(P.was, 0, wasBytes, stream);
  hipMemsetAsync(P.c2ph, 0, 16L * 256 * 4, stream);
  hipMemsetAsync(P.c2psum, 0, 16L * 1024 * 4, stream);

  // precompute
  k_pack<<<dim3(7680, 1, 14), 256, 0, stream>>>(P);
  k_featT<<<dim3(16384), 256, 0, stream>>>(P);
  k_featproj<<<dim3(128, 4, 2), 256, 0, stream>>>(P);
  k_avg<<<dim3(16), 256, 0, stream>>>(P);
  k_init<<<dim3(16), 256, 0, stream>>>(P);

  // sequential decode
  for (int i = 0; i < Tn; ++i) {
    k_gru1<<<dim3(2), 256, 0, stream>>>(P, i);
    k_query<<<dim3(2), 256, 0, stream>>>(P);
    k_energy<<<dim3(16, 8, 2), 256, 0, stream>>>(P, i);
    k_softmax<<<dim3(16, 2), 256, 0, stream>>>(P, i);
    k_ctx<<<dim3(16, 86, 2), 256, 0, stream>>>(P, i);
    k_gru2<<<dim3(2), 256, 0, stream>>>(P, i);
    k_proj<<<dim3(2), 256, 0, stream>>>(P, i);
  }
}
